// CantorMultiheadFusion_4793183502454
// MI455X (gfx1250) — matrix operand order repaired and checked by reading, not compiled, not run on hardware
//
#include <hip/hip_runtime.h>
#include <hip/hip_bf16.h>

// ---------------------------------------------------------------------------
// CantorMultiheadFusion for MI455X (gfx1250, wave32, WMMA + TDM)
//
// Pipeline:
//   1) cast x, Wq, Wk, Wv, Wo  f32 -> bf16   (workspace)
//   2) q/k/v = bf16 NT-GEMM (x * W^T): WMMA 16x16x32 bf16, f32 accum,
//      tiles DMA'd global->LDS by the Tensor Data Mover (double-buffered)
//   3) sliding-window (K=64) attention per 16-query block, WMMA for
//      Q*K^T and attn*V, softmax in registers (lane-half reductions)
//   4) out = ctx * Wo^T + bo  (f32 output)
// ---------------------------------------------------------------------------

typedef __bf16 bf16_t;
typedef __attribute__((ext_vector_type(4)))  __bf16 bf16x4;
typedef __attribute__((ext_vector_type(8)))  __bf16 bf16x8;
typedef __attribute__((ext_vector_type(16))) __bf16 v16bf;
typedef __attribute__((ext_vector_type(8)))  float  v8f;
typedef __attribute__((ext_vector_type(4)))  unsigned int u32x4;
typedef __attribute__((ext_vector_type(8)))  int         i32x8;
typedef __attribute__((ext_vector_type(4)))  int         i32x4;

#if __has_builtin(__builtin_amdgcn_tensor_load_to_lds) && \
    __has_builtin(__builtin_amdgcn_s_wait_tensorcnt)
#define HAVE_TDM 1
#else
#define HAVE_TDM 0
#endif

#define WMMA_BF16(a, b, c) \
  __builtin_amdgcn_wmma_f32_16x16x32_bf16(false, (a), false, (b), (short)0, (c), false, false)

static __device__ inline v16bf make_v16(bf16x8 lo, bf16x8 hi) {
  return __builtin_shufflevector(lo, hi, 0, 1, 2, 3, 4, 5, 6, 7,
                                         8, 9, 10, 11, 12, 13, 14, 15);
}

// ---------------------------------------------------------------------------
// f32 -> bf16 conversion (vectorized x4)
// ---------------------------------------------------------------------------
__global__ __launch_bounds__(256) void f32_to_bf16_kernel(
    const float* __restrict__ src, bf16_t* __restrict__ dst, int n) {
  int i = (blockIdx.x * blockDim.x + threadIdx.x) * 4;
  if (i < n) {
    float4 f = *reinterpret_cast<const float4*>(src + i);
    bf16x4 o;
    o[0] = (bf16_t)f.x; o[1] = (bf16_t)f.y; o[2] = (bf16_t)f.z; o[3] = (bf16_t)f.w;
    *reinterpret_cast<bf16x4*>(dst + i) = o;
  }
}

// ---------------------------------------------------------------------------
// WMMA fragment loads from LDS tiles (row stride 32 bf16)
// ---------------------------------------------------------------------------
// A-fragment (16x32, 16-bit): lane m = lane&15; half = lane>>4 picks the
// K-interleave: v0..3 = K[half*8 .. +8), v4..7 = K[16+half*8 .. +8)
static __device__ inline v16bf frag_a_lds(const bf16_t* base, int row0) {
  int lane = threadIdx.x & 31;
  int m = lane & 15, half = lane >> 4;
  const bf16_t* r = base + (row0 + m) * 32;
  return make_v16(*reinterpret_cast<const bf16x8*>(r + half * 8),
                  *reinterpret_cast<const bf16x8*>(r + 16 + half * 8));
}
// B-fragment (32x16, 16-bit): lane n = lane&15 holds column n of B
// (= row n of B^T = contiguous W row); same K-interleave as the A-fragment.
static __device__ inline v16bf frag_b_lds(const bf16_t* base, int col0) {
  int lane = threadIdx.x & 31;
  int n = lane & 15, half = lane >> 4;
  const bf16_t* r = base + (col0 + n) * 32;
  return make_v16(*reinterpret_cast<const bf16x8*>(r + half * 8),
                  *reinterpret_cast<const bf16x8*>(r + 16 + half * 8));
}

#if HAVE_TDM
// ---------------------------------------------------------------------------
// Tensor Data Mover: DMA a [rows x 32] bf16 tile (row stride 1024 elements)
// from global memory into LDS. D# per CDNA5 ISA ch.8:
//   group0: count=1 | lds_addr | global_addr[56:0] | type=2
//   group1: data_size=1(2B), tensor_dim0=1024, tensor_dim1=tensor_rows,
//           tile_dim0=32, tile_dim1=rows, tensor_dim0_stride=1024
//   groups 2/3 (and the trailing i32x8 descriptor words): zero (2D tensor)
// Generic LDS pointers carry the LDS byte offset in their low 32 bits.
// ---------------------------------------------------------------------------
static __device__ inline void tdm_load_tile32(const bf16_t* gsrc,
                                              const bf16_t* lds_dst,
                                              unsigned rows,
                                              unsigned tensor_rows) {
  unsigned long long ga = (unsigned long long)(uintptr_t)gsrc;
  unsigned lds_off = (unsigned)(uintptr_t)lds_dst;  // low 32 bits = LDS offset

  u32x4 g0;
  g0[0] = 1u;                                        // count=1, user D#
  g0[1] = lds_off;                                   // lds_addr (bytes)
  g0[2] = (unsigned)(ga & 0xffffffffu);              // global_addr[31:0]
  g0[3] = (unsigned)((ga >> 32) & 0x01ffffffu)       // global_addr[56:32]
          | (2u << 30);                              // type = 2 ("image")

  const unsigned td0 = 1024u;                        // tensor dim0 (elements)
  const unsigned td1 = tensor_rows;                  // tensor dim1 (rows)
  const unsigned tile0 = 32u, tile1 = rows, tile2 = 0u;
  const unsigned long long s0 = 1024ull;             // dim0 stride (elements)

  i32x8 g1;
  g1[0] = (int)(1u << 16);                           // data_size=1 (2 bytes)
  g1[1] = (int)((td0 & 0xffffu) << 16);              // tensor_dim0[15:0]
  g1[2] = (int)((td0 >> 16) | ((td1 & 0xffffu) << 16));
  g1[3] = (int)((td1 >> 16) | (tile0 << 16));        // tile_dim0
  g1[4] = (int)(tile1 | (tile2 << 16));              // tile_dim1, tile_dim2
  g1[5] = (int)(unsigned)(s0 & 0xffffffffu);         // dim0 stride lo
  g1[6] = (int)(unsigned)((s0 >> 32) & 0xffffu);     // stride hi | dim1str lo=0
  g1[7] = 0;

  i32x4 z4 = {0, 0, 0, 0};
  i32x8 z8 = {0, 0, 0, 0, 0, 0, 0, 0};
  __builtin_amdgcn_tensor_load_to_lds(g0, g1, z4, z4, z8, 0);
}
#endif  // HAVE_TDM

// ---------------------------------------------------------------------------
// NT GEMM: C[m,n] = sum_k A[m,k] * W[n,k] (+ bias[n])
// A: 4096 x 1024 bf16 row-major, W: 1024 x 1024 bf16 row-major.
// Workgroup: 256 threads (8 waves), 128x128 C tile, K staged 32 at a time.
// TDM path: wave 0 DMAs both tiles into the alternate LDS buffer while all
// waves compute from the current one (double-buffered, TENSORcnt-ordered).
// ---------------------------------------------------------------------------
__global__ __launch_bounds__(256) void gemm_bf16_nt_kernel(
    const bf16_t* __restrict__ A, const bf16_t* __restrict__ W,
    float* __restrict__ Cf, bf16_t* __restrict__ Cb,
    const float* __restrict__ bias) {
  __shared__ bf16_t As[2][128 * 32];
  __shared__ bf16_t Ws[2][128 * 32];

  const int tid  = threadIdx.x;
  const int wave = tid >> 5;
  const int bm   = blockIdx.y * 128;
  const int bn   = blockIdx.x * 128;
  const int wm0  = (wave >> 1) * 32;  // 0,32,64,96
  const int wn0  = (wave & 1) * 64;   // 0,64

  v8f acc[2][4] = {};

#if HAVE_TDM
  const bf16_t* Abase = A + (size_t)bm * 1024;
  const bf16_t* Wbase = W + (size_t)bn * 1024;
  const bool issuer = (wave == 0);

  if (issuer) {  // prologue: stage k0=0 into buffer 0
    tdm_load_tile32(Abase, &As[0][0], 128u, 4096u);
    tdm_load_tile32(Wbase, &Ws[0][0], 128u, 1024u);
  }

  for (int it = 0; it < 32; ++it) {
    const int buf = it & 1;
    if (issuer) {
      if (it + 1 < 32) {  // stage next K block into the other buffer
        const int kn = (it + 1) * 32;
        tdm_load_tile32(Abase + kn, &As[buf ^ 1][0], 128u, 4096u);
        tdm_load_tile32(Wbase + kn, &Ws[buf ^ 1][0], 128u, 1024u);
        __builtin_amdgcn_s_wait_tensorcnt(2);  // current buffer complete
      } else {
        __builtin_amdgcn_s_wait_tensorcnt(0);
      }
    }
    __syncthreads();

    v16bf a0 = frag_a_lds(As[buf], wm0);
    v16bf a1 = frag_a_lds(As[buf], wm0 + 16);
#pragma unroll
    for (int nt = 0; nt < 4; ++nt) {
      v16bf bw = frag_b_lds(Ws[buf], wn0 + nt * 16);
      acc[0][nt] = WMMA_BF16(a0, bw, acc[0][nt]);
      acc[1][nt] = WMMA_BF16(a1, bw, acc[1][nt]);
    }
    __syncthreads();  // buffer may be overwritten next iteration
  }
#else
  // Fallback: cooperative VGPR staging (global_load_b128 -> ds_store_b128)
  for (int k0 = 0; k0 < 1024; k0 += 32) {
#pragma unroll
    for (int c = 0; c < 2; ++c) {
      int chunk = tid + c * 256;
      int r  = chunk >> 2;
      int cc = (chunk & 3) * 8;
      *reinterpret_cast<bf16x8*>(&As[0][r * 32 + cc]) =
          *reinterpret_cast<const bf16x8*>(&A[(size_t)(bm + r) * 1024 + k0 + cc]);
      *reinterpret_cast<bf16x8*>(&Ws[0][r * 32 + cc]) =
          *reinterpret_cast<const bf16x8*>(&W[(size_t)(bn + r) * 1024 + k0 + cc]);
    }
    __syncthreads();
    v16bf a0 = frag_a_lds(As[0], wm0);
    v16bf a1 = frag_a_lds(As[0], wm0 + 16);
#pragma unroll
    for (int nt = 0; nt < 4; ++nt) {
      v16bf bw = frag_b_lds(Ws[0], wn0 + nt * 16);
      acc[0][nt] = WMMA_BF16(a0, bw, acc[0][nt]);
      acc[1][nt] = WMMA_BF16(a1, bw, acc[1][nt]);
    }
    __syncthreads();
  }
#endif

  // C layout: VGPR i -> row i + (lane>>4)*8, col = lane&15
  const int lane = tid & 31, half = lane >> 4, ln = lane & 15;
#pragma unroll
  for (int mt = 0; mt < 2; ++mt) {
#pragma unroll
    for (int nt = 0; nt < 4; ++nt) {
      int col = bn + wn0 + nt * 16 + ln;
      float bv = bias ? bias[col] : 0.0f;
#pragma unroll
      for (int i = 0; i < 8; ++i) {
        int row = bm + wm0 + mt * 16 + half * 8 + i;
        float v = acc[mt][nt][i] + bv;
        if (Cf) Cf[(size_t)row * 1024 + col] = v;
        if (Cb) Cb[(size_t)row * 1024 + col] = (bf16_t)v;
      }
    }
  }
}

// ---------------------------------------------------------------------------
// Sliding-window attention: one wave per (b, h, 16-query block).
// Queries [m0, m0+16); keys needed: [m0-32, m0+47) -> 5 key tiles of 16.
// scores = Q(16x64) * K^T        : 5 tiles x 2 WMMA = 10 WMMA
// out    = softmax(scores) * V   : 3 j-chunks x 4 d-tiles = 12 WMMA
// ---------------------------------------------------------------------------
__global__ __launch_bounds__(32) void window_attn_kernel(
    const bf16_t* __restrict__ q, const bf16_t* __restrict__ k,
    const bf16_t* __restrict__ v, bf16_t* __restrict__ ctx) {
  const int S = 2048;
  __shared__ bf16_t Vt[64 * 96];  // V^T tile: [d][j], j padded 80 -> 96
  __shared__ bf16_t At[16 * 96];  // attn tile: [m][j]

  const int bid  = blockIdx.x;
  const int sblk = bid & 127;
  const int h    = (bid >> 7) & 15;
  const int b    = bid >> 11;
  const int m0   = sblk * 16;
  const int kt0  = m0 - 32;

  const int lane = threadIdx.x;
  const int half = lane >> 4, ln = lane & 15;
  const size_t rowbase = (size_t)b * S;
  const int colh = h * 64;

  // zero attn tile (j >= 80 pad must be 0 for the attn*V contraction)
  for (int i = lane; i < 16 * 96 / 2; i += 32)
    reinterpret_cast<unsigned int*>(At)[i] = 0u;

  // stage V^T into LDS (rows clamped at sequence edges; pad cols zeroed)
  for (int j = 0; j < 96; ++j) {
    int praw = kt0 + j;
    int p = praw < 0 ? 0 : (praw > S - 1 ? S - 1 : praw);
#pragma unroll
    for (int c = 0; c < 2; ++c) {
      int d = lane + c * 32;
      bf16_t val = (j < 80) ? v[(rowbase + p) * 1024 + colh + d] : (bf16_t)0.0f;
      Vt[d * 96 + j] = val;
    }
  }
  __syncthreads();

  // Q fragments (A-matrix, 16x32 per chunk over d)
  v16bf aq[2];
#pragma unroll
  for (int ch = 0; ch < 2; ++ch) {
    const bf16_t* qr = q + (rowbase + m0 + ln) * 1024 + colh + ch * 32;
    aq[ch] = make_v16(*reinterpret_cast<const bf16x8*>(qr + half * 8),
                      *reinterpret_cast<const bf16x8*>(qr + 16 + half * 8));
  }

  // scores: 5 key tiles
  v8f sc[5];
#pragma unroll
  for (int t = 0; t < 5; ++t) {
    sc[t] = (v8f){0, 0, 0, 0, 0, 0, 0, 0};
    int praw = kt0 + t * 16 + ln;
    int p = praw < 0 ? 0 : (praw > S - 1 ? S - 1 : praw);
    const bf16_t* kr = k + (rowbase + p) * 1024 + colh;
#pragma unroll
    for (int ch = 0; ch < 2; ++ch) {
      v16bf bk = make_v16(
          *reinterpret_cast<const bf16x8*>(kr + ch * 32 + half * 8),
          *reinterpret_cast<const bf16x8*>(kr + ch * 32 + 16 + half * 8));
      sc[t] = WMMA_BF16(aq[ch], bk, sc[t]);
    }
  }

  // band mask + scale, then row-wise softmax.
  // C layout: lane ln = key col within tile, VGPR i -> query row half*8 + i.
  const float scale = 0.125f;  // 1/sqrt(64)
  float rowmax[8];
#pragma unroll
  for (int i = 0; i < 8; ++i) rowmax[i] = -1e30f;
#pragma unroll
  for (int t = 0; t < 5; ++t) {
    int praw = kt0 + t * 16 + ln;
#pragma unroll
    for (int i = 0; i < 8; ++i) {
      int qs  = m0 + half * 8 + i;
      int rel = praw - qs + 32;
      bool ok = (rel >= 0) && (rel < 64) && (praw >= 0) && (praw < S);
      float sv = ok ? sc[t][i] * scale : -1e30f;
      sc[t][i] = sv;
      rowmax[i] = fmaxf(rowmax[i], sv);
    }
  }
#pragma unroll
  for (int i = 0; i < 8; ++i)
    for (int off = 1; off < 16; off <<= 1)
      rowmax[i] = fmaxf(rowmax[i], __shfl_xor(rowmax[i], off, 16));

  float rowsum[8] = {};
#pragma unroll
  for (int t = 0; t < 5; ++t)
#pragma unroll
    for (int i = 0; i < 8; ++i) {
      float e = __expf(sc[t][i] - rowmax[i]);
      sc[t][i] = e;
      rowsum[i] += e;
    }
#pragma unroll
  for (int i = 0; i < 8; ++i) {
    for (int off = 1; off < 16; off <<= 1)
      rowsum[i] += __shfl_xor(rowsum[i], off, 16);
    rowsum[i] = 1.0f / rowsum[i];
  }

  // attn -> LDS (bf16) in A-matrix-friendly row-major [m][j]
#pragma unroll
  for (int t = 0; t < 5; ++t)
#pragma unroll
    for (int i = 0; i < 8; ++i) {
      int m = half * 8 + i;
      At[m * 96 + t * 16 + ln] = (bf16_t)(sc[t][i] * rowsum[i]);
    }
  __syncthreads();

  // out = attn(16x96) * V(96x64)
  v8f o[4] = {};
#pragma unroll
  for (int jc = 0; jc < 3; ++jc) {
    const bf16_t* ar = &At[ln * 96 + jc * 32];
    v16bf af = make_v16(*reinterpret_cast<const bf16x8*>(ar + half * 8),
                        *reinterpret_cast<const bf16x8*>(ar + 16 + half * 8));
#pragma unroll
    for (int dt = 0; dt < 4; ++dt) {
      const bf16_t* vr = &Vt[(dt * 16 + ln) * 96 + jc * 32];
      v16bf bv = make_v16(*reinterpret_cast<const bf16x8*>(vr + half * 8),
                          *reinterpret_cast<const bf16x8*>(vr + 16 + half * 8));
      o[dt] = WMMA_BF16(af, bv, o[dt]);
    }
  }

  // store context (bf16) for the Wo projection
#pragma unroll
  for (int dt = 0; dt < 4; ++dt)
#pragma unroll
    for (int i = 0; i < 8; ++i) {
      int row = m0 + half * 8 + i;
      ctx[(rowbase + row) * 1024 + colh + dt * 16 + ln] = (bf16_t)o[dt][i];
    }
}

// ---------------------------------------------------------------------------
// launch
// ---------------------------------------------------------------------------
extern "C" void kernel_launch(void* const* d_in, const int* in_sizes, int n_in,
                              void* d_out, int out_size, void* d_ws, size_t ws_size,
                              hipStream_t stream) {
  const float* x  = (const float*)d_in[0];
  // d_in[1] = cantor_coords: unused by the reference computation
  const float* Wq = (const float*)d_in[2];
  const float* Wk = (const float*)d_in[3];
  const float* Wv = (const float*)d_in[4];
  const float* Wo = (const float*)d_in[5];
  const float* bo = (const float*)d_in[6];

  char* ws = (char*)d_ws;
  const size_t MB = 1024ull * 1024ull;
  bf16_t* xb  = (bf16_t*)(ws + 0 * MB);   // 4096x1024 bf16 (8 MiB)
  bf16_t* wqb = (bf16_t*)(ws + 8 * MB);   // 1024x1024 bf16 (2 MiB each)
  bf16_t* wkb = (bf16_t*)(ws + 10 * MB);
  bf16_t* wvb = (bf16_t*)(ws + 12 * MB);
  bf16_t* wob = (bf16_t*)(ws + 14 * MB);
  bf16_t* qb  = (bf16_t*)(ws + 16 * MB);  // 8 MiB each
  bf16_t* kb  = (bf16_t*)(ws + 24 * MB);
  bf16_t* vb  = (bf16_t*)(ws + 32 * MB);
  bf16_t* ctx = (bf16_t*)(ws + 40 * MB);  // total 48 MiB workspace

  const int nx = 4096 * 1024, nw = 1024 * 1024;
  f32_to_bf16_kernel<<<nx / 4 / 256, 256, 0, stream>>>(x, xb, nx);
  f32_to_bf16_kernel<<<nw / 4 / 256, 256, 0, stream>>>(Wq, wqb, nw);
  f32_to_bf16_kernel<<<nw / 4 / 256, 256, 0, stream>>>(Wk, wkb, nw);
  f32_to_bf16_kernel<<<nw / 4 / 256, 256, 0, stream>>>(Wv, wvb, nw);
  f32_to_bf16_kernel<<<nw / 4 / 256, 256, 0, stream>>>(Wo, wob, nw);

  dim3 gg(1024 / 128, 4096 / 128);  // (N tiles, M tiles)
  gemm_bf16_nt_kernel<<<gg, 256, 0, stream>>>(xb, wqb, nullptr, qb, nullptr);
  gemm_bf16_nt_kernel<<<gg, 256, 0, stream>>>(xb, wkb, nullptr, kb, nullptr);
  gemm_bf16_nt_kernel<<<gg, 256, 0, stream>>>(xb, wvb, nullptr, vb, nullptr);

  window_attn_kernel<<<2 * 16 * (2048 / 16), 32, 0, stream>>>(qb, kb, vb, ctx);

  gemm_bf16_nt_kernel<<<gg, 256, 0, stream>>>(ctx, wob, (float*)d_out, nullptr, bo);
}
